// GraphRecurNetGate_8916352106978
// MI455X (gfx1250) — compile-verified, matrix operand order checked by reading, not run on hardware
//
#include <hip/hip_runtime.h>
#include <hip/hip_bf16.h>
#include <math.h>

// ---------------------------------------------------------------------------
// GraphRecurNetGate for MI455X (gfx1250, wave32, WMMA 16x16x32 f16).
// Fused GCN(2-layer, B=128 graph) + LSTM(T=365) kernel: grid = N = 32 blocks,
// 256 threads = 8 waves, wave w owns rows [16w,16w+16) of the 128-basin tile.
// All matmuls use v_wmma_f32_16x16x32_f16, f32 accumulate. Elementwise
// nonlinearities are branch-free fast paths on v_exp_f32 / v_rcp_f32.
// Next-timestep input is prefetched (global_prefetch_b8) to hide HBM latency.
// ---------------------------------------------------------------------------

typedef _Float16 h16;
typedef __attribute__((ext_vector_type(16))) _Float16 v16h;
typedef __attribute__((ext_vector_type(8)))  _Float16 v8h;
typedef __attribute__((ext_vector_type(4)))  _Float16 v4h;
typedef __attribute__((ext_vector_type(8)))  float    v8f;

#define NN     32
#define TT     365
#define BB     128
#define DENSE  32
#define SPARSE 16
#define CCH    (DENSE + SPARSE)   // 48
#define PP     32
#define HH     32

// ---- branch-free fast math (native v_exp_f32 / v_rcp_f32 / v_tanh_f32) ----
__device__ __forceinline__ float fexp(float x) {
  return __builtin_amdgcn_exp2f(x * 1.44269504088896340736f);  // v_mul + v_exp
}
__device__ __forceinline__ float frcp(float x) {
  return __builtin_amdgcn_rcpf(x);                              // v_rcp
}
// erf via Abramowitz-Stegun 7.1.26 (|err| <= 1.5e-7), branch-free.
__device__ __forceinline__ float erf_fast(float x) {
  float ax = fabsf(x);
  float t  = frcp(fmaf(0.3275911f, ax, 1.0f));
  float y  = t * fmaf(t, fmaf(t, fmaf(t, fmaf(t, 1.061405429f, -1.453152027f),
                                      1.421413741f), -0.284496736f), 0.254829592f);
  float r  = fmaf(-y, fexp(-ax * ax), 1.0f);
  return copysignf(r, x);
}
__device__ __forceinline__ float gelu_f(float x) {
  return 0.5f * x * (1.0f + erf_fast(x * 0.70710678118654752f));
}
__device__ __forceinline__ float sig_f(float x) {
  return frcp(1.0f + fexp(-x));
}
__device__ __forceinline__ float tanh_f(float x) {
#if __has_builtin(__builtin_amdgcn_tanhf)
  return __builtin_amdgcn_tanhf(x);                             // v_tanh_f32
#else
  float e = fexp(-2.0f * fabsf(x));
  float r = (1.0f - e) * frcp(1.0f + e);
  return copysignf(r, x);
#endif
}
__device__ __forceinline__ v8f splat8(float x) { v8f v = {x,x,x,x,x,x,x,x}; return v; }

// Fragment loader: 32 contiguous halves at `base` hold one row (A) or one
// column (B) of K=32. Lanes 0-15 take K=[0..7]+[16..23], lanes 16-31 take
// K=[8..15]+[24..31] per the CDNA5 16-bit A/B VGPR layout.
__device__ __forceinline__ v16h frag_ld(const h16* base, int halfsel) {
  const v8h* p = (const v8h*)base;
  v8h lo = p[halfsel];
  v8h hi = p[halfsel + 2];
  return __builtin_shufflevector(lo, hi, 0,1,2,3,4,5,6,7,8,9,10,11,12,13,14,15);
}

__device__ __forceinline__ v8f wmma_f16(v16h a, v16h b, v8f c) {
  return __builtin_amdgcn_wmma_f32_16x16x32_f16(false, a, false, b, (short)0, c,
                                                false, false);
}

// ---------------------------------------------------------------------------
// Prep: normalized adjacency (f16) + f16 weight repacks (B-operand col-major).
// ---------------------------------------------------------------------------
__global__ __launch_bounds__(256) void k_prep(
    const float* __restrict__ adj, const float* __restrict__ pd_W,
    const float* __restrict__ g1_W, const float* __restrict__ g2_W,
    const float* __restrict__ l_Wih, const float* __restrict__ l_Whh,
    h16* __restrict__ adjn, h16* __restrict__ pdWT, h16* __restrict__ g1WT,
    h16* __restrict__ g2WT, h16* __restrict__ wihB, h16* __restrict__ whhB) {
  __shared__ float d[BB];
  int tid = threadIdx.x;
  if (tid < BB) {
    float s = 0.f;
    for (int v = 0; v < BB; ++v) s += adj[tid * BB + v];
    d[tid] = rsqrtf(fmaxf(s, 1.0f));
  }
  __syncthreads();
  for (int i = tid; i < BB * BB; i += 256) {
    int u = i >> 7, v = i & 127;
    adjn[i] = (h16)(d[u] * adj[i] * d[v]);
  }
  for (int i = tid; i < 32 * 32; i += 256) {  // [n][k] = W[k][n]
    int nc = i >> 5, k = i & 31;
    pdWT[i] = (h16)pd_W[k * 32 + nc];
    g1WT[i] = (h16)g1_W[k * 32 + nc];
    g2WT[i] = (h16)g2_W[k * 32 + nc];
  }
  for (int i = tid; i < 128 * 32; i += 256) {  // l_W*[nn][k] already col-major for h @ W.T
    wihB[i] = (h16)l_Wih[i];
    whhB[i] = (h16)l_Whh[i];
  }
}

// ---------------------------------------------------------------------------
// Fused GCN + LSTM. One block per n. LDS = 32KB adjn + 3 x 8KB tiles = 56KB.
// ---------------------------------------------------------------------------
__global__ __launch_bounds__(256, 1) void k_fused(
    const float* __restrict__ input, const float* __restrict__ pd_b,
    const float* __restrict__ g1_b, const float* __restrict__ g2_b,
    const float* __restrict__ l_bih, const float* __restrict__ l_bhh,
    const h16* __restrict__ adjn, const h16* __restrict__ pdWT,
    const h16* __restrict__ g1WT, const h16* __restrict__ g2WT,
    const h16* __restrict__ wihB, const h16* __restrict__ whhB,
    float* __restrict__ hlast) {
  __shared__ __align__(16) h16 sAdj[BB * BB];  // row-major adjn
  __shared__ __align__(16) h16 sXB[BB * HH];   // X/G stage: row-major ld32 OR col-major ld128
  __shared__ __align__(16) h16 sY[BB * HH];    // activation stage, row-major ld32
  __shared__ __align__(16) h16 sHH[BB * HH];   // LSTM hidden state, row-major ld32

  const int tid  = threadIdx.x;
  const int lane = tid & 31;
  const int w    = tid >> 5;
  const int half = lane >> 4;
  const int l15  = lane & 15;
  const int m0   = w * 16;
  const int n    = blockIdx.x;

  {  // LDS init
    uint4 z = make_uint4(0u, 0u, 0u, 0u);
    const uint4* ga = (const uint4*)adjn;
    uint4* la = (uint4*)sAdj;
    for (int i = tid; i < (BB * BB * 2 / 16); i += 256) la[i] = ga[i];
    uint4* lh = (uint4*)sHH;
    for (int i = tid; i < (BB * HH * 2 / 16); i += 256) lh[i] = z;
  }

  // Loop-invariant B fragments + biases in registers.
  v16h bpd[2], bg1[2], bg2[2];
  float bias_pd[2], bias_g1[2], bias_g2[2], bias_lstm[8];
#pragma unroll
  for (int nt = 0; nt < 2; ++nt) {
    bpd[nt] = frag_ld(&pdWT[(nt * 16 + l15) * 32], half);
    bg1[nt] = frag_ld(&g1WT[(nt * 16 + l15) * 32], half);
    bg2[nt] = frag_ld(&g2WT[(nt * 16 + l15) * 32], half);
    bias_pd[nt] = pd_b[nt * 16 + l15];
    bias_g1[nt] = g1_b[nt * 16 + l15];
    bias_g2[nt] = g2_b[nt * 16 + l15];
  }
#pragma unroll
  for (int nt = 0; nt < 8; ++nt)
    bias_lstm[nt] = l_bih[nt * 16 + l15] + l_bhh[nt * 16 + l15];

  float cc[2][8];
#pragma unroll
  for (int a = 0; a < 2; ++a)
#pragma unroll
    for (int r = 0; r < 8; ++r) cc[a][r] = 0.f;

  // Prefetch first slice.
  {
    const char* p0 = (const char*)(input + ((size_t)n * TT) * BB * CCH);
    if (tid < 192) __builtin_prefetch(p0 + tid * 128, 0, 1);
  }

  for (int t = 0; t < TT; ++t) {
    __syncthreads();  // S0
    // Stage dense features (f32 -> f16), row-major ld32. 4 x b128 loads +
    // 4 x b64 LDS stores per thread (rows are 192B, 16B-aligned chunks).
    const float* xin = input + (((size_t)n * TT + t) * BB) * CCH;
    for (int i = tid; i < BB * (DENSE / 4); i += 256) {
      int b = i >> 3, c4 = (i & 7) * 4;
      const float4 v = *(const float4*)&xin[b * CCH + c4];
      v4h hv = { (h16)v.x, (h16)v.y, (h16)v.z, (h16)v.w };
      *(v4h*)&sXB[(b << 5) + c4] = hv;
    }
    // Prefetch next slice (24KB = 192 x 128B lines) while this step computes.
    if (t + 1 < TT && tid < 192) {
      const char* pn = (const char*)(xin + (size_t)BB * CCH);
      __builtin_prefetch(pn + tid * 128, 0, 1);
    }
    __syncthreads();  // S1

    // GEMM1: P0 = gelu(X @ pd_W + pd_b) -> sY
    {
      v16h a = frag_ld(&sXB[(m0 + l15) * 32], half);
#pragma unroll
      for (int nt = 0; nt < 2; ++nt) {
        v8f acc = splat8(bias_pd[nt]);
        acc = wmma_f16(a, bpd[nt], acc);
#pragma unroll
        for (int r = 0; r < 8; ++r)
          sY[(m0 + half * 8 + r) * 32 + nt * 16 + l15] = (h16)gelu_f(acc[r]);
      }
    }
    __syncthreads();  // S2

    // GEMM2: G1 = P0 @ g1_W -> sXB col-major [n][k], ld128
    {
      v16h a = frag_ld(&sY[(m0 + l15) * 32], half);
#pragma unroll
      for (int nt = 0; nt < 2; ++nt) {
        v8f acc = splat8(0.f);
        acc = wmma_f16(a, bg1[nt], acc);
#pragma unroll
        for (int r = 0; r < 8; ++r)
          sXB[(nt * 16 + l15) * 128 + m0 + half * 8 + r] = (h16)acc[r];
      }
    }
    __syncthreads();  // S3

    // GEMM3: H1 = gelu(adjn @ G1 + g1_b) -> sY   (K=128: 4-deep WMMA chain)
#pragma unroll
    for (int nt = 0; nt < 2; ++nt) {
      v8f acc = splat8(bias_g1[nt]);
#pragma unroll
      for (int kt = 0; kt < 4; ++kt) {
        v16h a = frag_ld(&sAdj[(m0 + l15) * 128 + kt * 32], half);
        v16h b = frag_ld(&sXB[(nt * 16 + l15) * 128 + kt * 32], half);
        acc = wmma_f16(a, b, acc);
      }
#pragma unroll
      for (int r = 0; r < 8; ++r)
        sY[(m0 + half * 8 + r) * 32 + nt * 16 + l15] = (h16)gelu_f(acc[r]);
    }
    __syncthreads();  // S4

    // GEMM4: G2 = H1 @ g2_W -> sXB col-major
    {
      v16h a = frag_ld(&sY[(m0 + l15) * 32], half);
#pragma unroll
      for (int nt = 0; nt < 2; ++nt) {
        v8f acc = splat8(0.f);
        acc = wmma_f16(a, bg2[nt], acc);
#pragma unroll
        for (int r = 0; r < 8; ++r)
          sXB[(nt * 16 + l15) * 128 + m0 + half * 8 + r] = (h16)acc[r];
      }
    }
    __syncthreads();  // S5

    // GEMM5: H2 = gelu(adjn @ G2 + g2_b) -> sY
#pragma unroll
    for (int nt = 0; nt < 2; ++nt) {
      v8f acc = splat8(bias_g2[nt]);
#pragma unroll
      for (int kt = 0; kt < 4; ++kt) {
        v16h a = frag_ld(&sAdj[(m0 + l15) * 128 + kt * 32], half);
        v16h b = frag_ld(&sXB[(nt * 16 + l15) * 128 + kt * 32], half);
        acc = wmma_f16(a, b, acc);
      }
#pragma unroll
      for (int r = 0; r < 8; ++r)
        sY[(m0 + half * 8 + r) * 32 + nt * 16 + l15] = (h16)gelu_f(acc[r]);
    }
    __syncthreads();  // S6

    // GEMM6 + LSTM step: g = H2 @ Wih.T + hh @ Whh.T + (bih+bhh); gates in-reg.
    {
      v16h aH = frag_ld(&sY[(m0 + l15) * 32], half);
      v16h aS = frag_ld(&sHH[(m0 + l15) * 32], half);
      v8f g[8];
#pragma unroll
      for (int nt = 0; nt < 8; ++nt) {
        v8f acc = splat8(bias_lstm[nt]);
        v16h bw = frag_ld(&wihB[(nt * 16 + l15) * 32], half);
        v16h bu = frag_ld(&whhB[(nt * 16 + l15) * 32], half);
        acc = wmma_f16(aH, bw, acc);
        acc = wmma_f16(aS, bu, acc);
        g[nt] = acc;
      }
      float hh[2][8];
#pragma unroll
      for (int nt = 0; nt < 2; ++nt)
#pragma unroll
        for (int r = 0; r < 8; ++r) {
          float iv = g[nt][r], fv = g[nt + 2][r], gv = g[nt + 4][r], ov = g[nt + 6][r];
          float c = sig_f(fv) * cc[nt][r] + sig_f(iv) * tanh_f(gv);
          cc[nt][r] = c;
          hh[nt][r] = sig_f(ov) * tanh_f(c);
        }
      __syncthreads();  // S7: all waves finished reading sHH
#pragma unroll
      for (int nt = 0; nt < 2; ++nt)
#pragma unroll
        for (int r = 0; r < 8; ++r)
          sHH[(m0 + half * 8 + r) * 32 + nt * 16 + l15] = (h16)hh[nt][r];
    }
  }
  __syncthreads();
  for (int i = tid; i < BB * HH; i += 256)
    hlast[(size_t)n * BB * HH + i] = (float)sHH[i];
}

// ---------------------------------------------------------------------------
// Tail: gather + sparse MLP + gates + batchnorm + readout (N=32, tiny).
// ---------------------------------------------------------------------------
__global__ __launch_bounds__(64, 1) void k_tail(
    const float* __restrict__ input, const int* __restrict__ basin_idx,
    const float* __restrict__ hlast, const float* __restrict__ ps_W,
    const float* __restrict__ ps_b, const float* __restrict__ sp_W1,
    const float* __restrict__ sp_b1, const float* __restrict__ sp_W2,
    const float* __restrict__ sp_b2, const float* __restrict__ gd_W1,
    const float* __restrict__ gd_b1, const float* __restrict__ gd_W2,
    const float* __restrict__ gd_b2, const float* __restrict__ gs_W1,
    const float* __restrict__ gs_b1, const float* __restrict__ gs_W2,
    const float* __restrict__ gs_b2, const float* __restrict__ bn_g,
    const float* __restrict__ bn_b, const float* __restrict__ r_W,
    const float* __restrict__ r_b, float* __restrict__ out) {
  __shared__ float sh[NN * 64];
  int tid = threadIdx.x;
  if (tid < NN) {
    int bidx = basin_idx[tid];
    float xs[32];
    for (int h = 0; h < 32; ++h) xs[h] = hlast[((size_t)tid * BB + bidx) * HH + h];
    float sv[16];
    size_t base = (((size_t)tid * TT + (TT - 1)) * BB + bidx) * CCH + DENSE;
    for (int j = 0; j < 16; ++j) sv[j] = input[base + j];
    float t1[32], t2[32], sfin[32], u[64];
    for (int p = 0; p < 32; ++p) {
      float a = ps_b[p];
      for (int j = 0; j < 16; ++j) a += sv[j] * ps_W[j * 32 + p];
      t1[p] = gelu_f(a);
    }
    for (int h = 0; h < 32; ++h) {
      float a = sp_b1[h];
      for (int p = 0; p < 32; ++p) a += t1[p] * sp_W1[p * 32 + h];
      t2[h] = gelu_f(a);
    }
    for (int h = 0; h < 32; ++h) {
      float a = sp_b2[h];
      for (int p = 0; p < 32; ++p) a += t2[p] * sp_W2[p * 32 + h];
      sfin[h] = gelu_f(a);
    }
    for (int e = 0; e < 64; ++e) {
      float a = gd_b1[e];
      for (int h = 0; h < 32; ++h) a += xs[h] * gd_W1[h * 64 + e];
      u[e] = gelu_f(a);
    }
    for (int h = 0; h < 32; ++h) {
      float a = gd_b2[h];
      for (int e = 0; e < 64; ++e) a += u[e] * gd_W2[e * 32 + h];
      sh[tid * 64 + h] = xs[h] * sig_f(a);
    }
    for (int e = 0; e < 64; ++e) {
      float a = gs_b1[e];
      for (int h = 0; h < 32; ++h) a += xs[h] * gs_W1[h * 64 + e];
      u[e] = gelu_f(a);
    }
    for (int h = 0; h < 32; ++h) {
      float a = gs_b2[h];
      for (int e = 0; e < 64; ++e) a += u[e] * gs_W2[e * 32 + h];
      sh[tid * 64 + 32 + h] = sfin[h] * sig_f(a);
    }
  }
  __syncthreads();
  if (tid < 64) {
    float mu = 0.f;
    for (int i = 0; i < NN; ++i) mu += sh[i * 64 + tid];
    mu *= (1.0f / NN);
    float var = 0.f;
    for (int i = 0; i < NN; ++i) {
      float d = sh[i * 64 + tid] - mu;
      var += d * d;
    }
    var *= (1.0f / NN);
    float sc = rsqrtf(var + 1e-5f) * bn_g[tid];
    float bo = bn_b[tid];
    for (int i = 0; i < NN; ++i) sh[i * 64 + tid] = (sh[i * 64 + tid] - mu) * sc + bo;
  }
  __syncthreads();
  if (tid < NN) {
    float a = r_b[0];
    for (int c = 0; c < 64; ++c) a += sh[tid * 64 + c] * r_W[c];
    out[tid] = a;
  }
}

// ---------------------------------------------------------------------------
extern "C" void kernel_launch(void* const* d_in, const int* in_sizes, int n_in,
                              void* d_out, int out_size, void* d_ws, size_t ws_size,
                              hipStream_t stream) {
  (void)in_sizes; (void)n_in; (void)out_size; (void)ws_size;
  const float* input = (const float*)d_in[0];
  const float* adj   = (const float*)d_in[1];
  const int* basin_idx = (const int*)d_in[2];
  const float* pd_W = (const float*)d_in[3];  const float* pd_b = (const float*)d_in[4];
  const float* ps_W = (const float*)d_in[5];  const float* ps_b = (const float*)d_in[6];
  const float* g1_W = (const float*)d_in[7];  const float* g1_b = (const float*)d_in[8];
  const float* g2_W = (const float*)d_in[9];  const float* g2_b = (const float*)d_in[10];
  const float* l_Wih = (const float*)d_in[11]; const float* l_Whh = (const float*)d_in[12];
  const float* l_bih = (const float*)d_in[13]; const float* l_bhh = (const float*)d_in[14];
  const float* gd_W1 = (const float*)d_in[15]; const float* gd_b1 = (const float*)d_in[16];
  const float* gd_W2 = (const float*)d_in[17]; const float* gd_b2 = (const float*)d_in[18];
  const float* gs_W1 = (const float*)d_in[19]; const float* gs_b1 = (const float*)d_in[20];
  const float* gs_W2 = (const float*)d_in[21]; const float* gs_b2 = (const float*)d_in[22];
  const float* sp_W1 = (const float*)d_in[23]; const float* sp_b1 = (const float*)d_in[24];
  const float* sp_W2 = (const float*)d_in[25]; const float* sp_b2 = (const float*)d_in[26];
  const float* bn_g = (const float*)d_in[27];  const float* bn_b = (const float*)d_in[28];
  const float* r_W = (const float*)d_in[29];   const float* r_b = (const float*)d_in[30];

  // Workspace carve (~0.6 MB total).
  char* ws = (char*)d_ws;
  h16* adjn = (h16*)ws;  ws += 128 * 128 * 2;
  h16* pdWT = (h16*)ws;  ws += 32 * 32 * 2;
  h16* g1WT = (h16*)ws;  ws += 32 * 32 * 2;
  h16* g2WT = (h16*)ws;  ws += 32 * 32 * 2;
  h16* wihB = (h16*)ws;  ws += 128 * 32 * 2;
  h16* whhB = (h16*)ws;  ws += 128 * 32 * 2;
  ws = (char*)(((uintptr_t)ws + 255) & ~(uintptr_t)255);
  float* hlast = (float*)ws;  // N*B*H f32 = 512 KB

  k_prep<<<1, 256, 0, stream>>>(adj, pd_W, g1_W, g2_W, l_Wih, l_Whh,
                                adjn, pdWT, g1WT, g2WT, wihB, whhB);
  k_fused<<<NN, 256, 0, stream>>>(input, pd_b, g1_b, g2_b, l_bih, l_bhh,
                                  adjn, pdWT, g1WT, g2WT, wihB, whhB, hlast);
  k_tail<<<1, 64, 0, stream>>>(input, basin_idx, hlast, ps_W, ps_b, sp_W1, sp_b1,
                               sp_W2, sp_b2, gd_W1, gd_b1, gd_W2, gd_b2, gs_W1,
                               gs_b1, gs_W2, gs_b2, bn_g, bn_b, r_W, r_b,
                               (float*)d_out);
}
